// MaskedMultiheadSelfAttentionLayer_24764781428985
// MI455X (gfx1250) — matrix operand order repaired and checked by reading, not compiled, not run on hardware
//
#include <hip/hip_runtime.h>
#include <hip/hip_bf16.h>

// ---------------- CDNA5 vector types for WMMA ----------------
typedef __attribute__((ext_vector_type(16))) __bf16 v16bf;
typedef __attribute__((ext_vector_type(8)))  __bf16 v8bf;
typedef __attribute__((ext_vector_type(4)))  __bf16 v4bf;
typedef __attribute__((ext_vector_type(8)))  float  v8f;

// Problem constants (reference file)
static constexpr int BATCH = 4;
static constexpr int SEQ   = 2048;
static constexpr int DIM   = 1024;   // model dim
static constexpr int EMB   = 1024;   // embed dim

// ---------------------------------------------------------------------------
// fp32 -> bf16 bulk convert (n multiple of 4)
// ---------------------------------------------------------------------------
__global__ __launch_bounds__(256) void cvt_f32_to_bf16(const float* __restrict__ src,
                                                       __bf16* __restrict__ dst, long n) {
  long i0     = ((long)blockIdx.x * blockDim.x + threadIdx.x) * 4;
  long stride = (long)gridDim.x * blockDim.x * 4;
  for (long i = i0; i < n; i += stride) {
    float4 f = *reinterpret_cast<const float4*>(src + i);
    v4bf o = { (__bf16)f.x, (__bf16)f.y, (__bf16)f.z, (__bf16)f.w };
    *reinterpret_cast<v4bf*>(dst + i) = o;
  }
}

// ---------------------------------------------------------------------------
// Shared WMMA-bf16 GEMM core: C[m,n] = sum_k A[m,k] * Bm[n,k]  (both row-major)
//
// Block tile 64(M) x 128(N), 256 threads = 8 wave32 arranged 2(Mwaves) x 4(Nwaves).
// Each wave computes a 32x32 sub-tile = four 16x16 f32 accumulators
// (2 A-frags x 2 B-frags -> 4 v_wmma_f32_16x16x32_bf16 per K-step).
//
// K stepped by 32 through DOUBLE-BUFFERED LDS filled with
// GLOBAL_LOAD_ASYNC_TO_LDS_B128 (ASYNCcnt-tracked direct memory->LDS copies),
// so stage k+1 streams in while stage k is computed; one barrier per K-step.
//
// MODE 0: scores epilogue  (val*scale + 1e-13, causal fill -1e-13) -> bf16
// MODE 1: plain bf16 store
// MODE 2: fp32 store + bias[n]
// ---------------------------------------------------------------------------
template<int MODE>
__global__ __launch_bounds__(256) void gemm_bf16_tt(
    const __bf16* __restrict__ A,  int lda, long aBatch,
    const __bf16* __restrict__ Bm, int ldb, long bBatch,
    void* __restrict__ Cv,         int ldc, long cBatch,
    int Kdim, float scale, const float* __restrict__ bias) {

  __shared__ alignas(64) __bf16 As[2][64][32];    // 2 x 4 KB
  __shared__ alignas(64) __bf16 Bs[2][128][32];   // 2 x 8 KB

  const int t    = threadIdx.x;
  const int lane = t & 31;
  const int wave = t >> 5;
  const int wr   = wave & 1;      // wave M tile (0..1)  -> rows 32*wr
  const int wc   = wave >> 1;     // wave N tile (0..3)  -> cols 32*wc
  const int half = lane >> 4;     // half-wave (K split)
  const int nl   = lane & 15;

  const int mBase = blockIdx.y * 64;
  const int nBase = blockIdx.x * 128;
  const int batch = blockIdx.z;
  A  += (size_t)batch * aBatch;
  Bm += (size_t)batch * bBatch;

  // cooperative async fill mapping:
  //   A panel 64x32 bf16  = 256 x b128  -> 1 chunk/thread
  //   B panel 128x32 bf16 = 512 x b128  -> 2 chunks/thread
  const int arow = t >> 2;         // 0..63
  const int ach  = (t & 3) * 8;    // element offset 0,8,16,24
  const int brow = t >> 1;         // 0..127
  const int bch  = (t & 1) * 16;   // element offset 0,16

  const __bf16* gA = A  + (size_t)(mBase + arow) * lda + ach;
  const __bf16* gB = Bm + (size_t)(nBase + brow) * ldb + bch;

  auto issue_stage = [&](int kb, int buf) {
    unsigned la  = (unsigned)(uintptr_t)&As[buf][arow][ach];
    unsigned lb  = (unsigned)(uintptr_t)&Bs[buf][brow][bch];
    unsigned long long pa  = (unsigned long long)(uintptr_t)(gA + kb);
    unsigned long long pb0 = (unsigned long long)(uintptr_t)(gB + kb);
    unsigned long long pb1 = (unsigned long long)(uintptr_t)(gB + kb + 8);
    asm volatile("global_load_async_to_lds_b128 %0, %1, off"
                 :: "v"(la), "v"(pa) : "memory");
    asm volatile("global_load_async_to_lds_b128 %0, %1, off"
                 :: "v"(lb), "v"(pb0) : "memory");
    asm volatile("global_load_async_to_lds_b128 %0, %1, off"
                 :: "v"(lb + 16), "v"(pb1) : "memory");
  };

  v8f acc[2][2] = {{{}, {}}, {{}, {}}};

  // prologue: fill buffer 0
  issue_stage(0, 0);
  asm volatile("s_wait_asynccnt 0x0" ::: "memory");
  __syncthreads();

  int buf = 0;
  for (int kb = 0; kb < Kdim; kb += 32) {
    const int nxt = buf ^ 1;
    if (kb + 32 < Kdim) issue_stage(kb + 32, nxt);   // async prefetch next stage

    // A fragments (16x32): lane = M row; half-wave 0 holds K 0..15,
    // half-wave 1 holds K 16..31 (same k order as B)
    v16bf af[2];
#pragma unroll
    for (int rr = 0; rr < 2; ++rr) {
      const int r = wr * 32 + rr * 16 + nl;
      v8bf alo = *reinterpret_cast<const v8bf*>(&As[buf][r][half * 16]);
      v8bf ahi = *reinterpret_cast<const v8bf*>(&As[buf][r][half * 16 + 8]);
#pragma unroll
      for (int e = 0; e < 8; ++e) { af[rr][e] = alo[e]; af[rr][e + 8] = ahi[e]; }
    }
    // B fragments (32x16): lane = N col; elements = K half*16 + e
    v16bf bfr[2];
#pragma unroll
    for (int cc = 0; cc < 2; ++cc)
      bfr[cc] = *reinterpret_cast<const v16bf*>(&Bs[buf][wc * 32 + cc * 16 + nl][half * 16]);

#pragma unroll
    for (int rr = 0; rr < 2; ++rr)
#pragma unroll
      for (int cc = 0; cc < 2; ++cc)
        acc[rr][cc] = __builtin_amdgcn_wmma_f32_16x16x32_bf16(
            false, af[rr], false, bfr[cc], (short)0, acc[rr][cc], false, false);

    asm volatile("s_wait_asynccnt 0x0" ::: "memory");  // next stage landed in LDS
    __syncthreads();                                    // everyone done with `buf`
    buf = nxt;
  }

  // C layout: lanes 0-15 -> M = vgpr, lanes 16-31 -> M = vgpr+8; N = lane&15
#pragma unroll
  for (int rr = 0; rr < 2; ++rr)
#pragma unroll
    for (int cc = 0; cc < 2; ++cc)
#pragma unroll
      for (int i = 0; i < 8; ++i) {
        const int m = mBase + wr * 32 + rr * 16 + half * 8 + i;
        const int n = nBase + wc * 32 + cc * 16 + nl;
        const float a = acc[rr][cc][i];
        if (MODE == 0) {
          __bf16* C = reinterpret_cast<__bf16*>(Cv) + (size_t)batch * cBatch;
          float v = (n > m) ? -1e-13f : a * scale + 1e-13f;
          C[(size_t)m * ldc + n] = (__bf16)v;
        } else if (MODE == 1) {
          __bf16* C = reinterpret_cast<__bf16*>(Cv) + (size_t)batch * cBatch;
          C[(size_t)m * ldc + n] = (__bf16)a;
        } else {
          float* C = reinterpret_cast<float*>(Cv) + (size_t)batch * cBatch;
          C[(size_t)m * ldc + n] = a + bias[n];
        }
      }
}

// ---------------------------------------------------------------------------
// In-place row softmax over bf16 scores: one block per row of length SEQ
// ---------------------------------------------------------------------------
__global__ __launch_bounds__(256) void softmax_rows(__bf16* __restrict__ P, int S) {
  __shared__ float red[256];
  const int t = threadIdx.x;
  __bf16* row = P + (size_t)blockIdx.x * S;

  float v[8];
  float mx = -3.0e38f;
#pragma unroll
  for (int i = 0; i < 8; ++i) {
    v[i] = (float)row[t + i * 256];
    mx = fmaxf(mx, v[i]);
  }
  red[t] = mx; __syncthreads();
  for (int s2 = 128; s2 > 0; s2 >>= 1) {
    if (t < s2) red[t] = fmaxf(red[t], red[t + s2]);
    __syncthreads();
  }
  const float rmax = red[0];
  __syncthreads();

  float sum = 0.f;
#pragma unroll
  for (int i = 0; i < 8; ++i) { v[i] = __expf(v[i] - rmax); sum += v[i]; }
  red[t] = sum; __syncthreads();
  for (int s2 = 128; s2 > 0; s2 >>= 1) {
    if (t < s2) red[t] += red[t + s2];
    __syncthreads();
  }
  const float inv = 1.0f / red[0];
#pragma unroll
  for (int i = 0; i < 8; ++i) row[t + i * 256] = (__bf16)(v[i] * inv);
}

// ---------------------------------------------------------------------------
extern "C" void kernel_launch(void* const* d_in, const int* in_sizes, int n_in,
                              void* d_out, int out_size, void* d_ws, size_t ws_size,
                              hipStream_t stream) {
  const float* Q = (const float*)d_in[0];   // [B,S,D]
  const float* K = (const float*)d_in[1];   // [B,S,D]
  const float* V = (const float*)d_in[2];   // [B,D,S]
  const float* W = (const float*)d_in[3];   // [E,D]
  const float* b = (const float*)d_in[4];   // [E]

  const long nQ = (long)BATCH * SEQ * DIM;  // 8,388,608
  const long nK = nQ;
  const long nV = nQ;
  const long nW = (long)EMB * DIM;          // 1,048,576

  // workspace layout (bf16 buffers), ~98 MB total
  char* ws = (char*)d_ws;
  __bf16* Qb  = (__bf16*)(ws);                                   // 16 MB
  __bf16* Kb  = (__bf16*)(ws + (size_t)nQ * 2);                  // 16 MB
  __bf16* Vb  = (__bf16*)(ws + (size_t)(nQ + nK) * 2);           // 16 MB
  __bf16* Wb  = (__bf16*)(ws + (size_t)(nQ + nK + nV) * 2);      //  2 MB
  __bf16* Pb  = (__bf16*)(ws + (size_t)(nQ + nK + nV + nW) * 2); // 32 MB scores/probs
  const long nP = (long)BATCH * SEQ * SEQ;                       // 16,777,216
  __bf16* AOb = (__bf16*)(ws + (size_t)(nQ + nK + nV + nW + nP) * 2); // 16 MB attn_out

  // 1) precision shift: fp32 -> bf16 once (compute-bound workload => WMMA bf16)
  cvt_f32_to_bf16<<<dim3((unsigned)((nQ / 4 + 255) / 256)), 256, 0, stream>>>(Q, Qb, nQ);
  cvt_f32_to_bf16<<<dim3((unsigned)((nK / 4 + 255) / 256)), 256, 0, stream>>>(K, Kb, nK);
  cvt_f32_to_bf16<<<dim3((unsigned)((nV / 4 + 255) / 256)), 256, 0, stream>>>(V, Vb, nV);
  cvt_f32_to_bf16<<<dim3((unsigned)((nW / 4 + 255) / 256)), 256, 0, stream>>>(W, Wb, nW);

  // 2) scores = Qb @ Kb^T / sqrt(D) + 1e-13, causal fill -1e-13  (M=N=2048, K=1024)
  gemm_bf16_tt<0><<<dim3(SEQ / 128, SEQ / 64, BATCH), 256, 0, stream>>>(
      Qb, DIM, (long)SEQ * DIM,
      Kb, DIM, (long)SEQ * DIM,
      Pb, SEQ, (long)SEQ * SEQ,
      DIM, 0.03125f /* 1/sqrt(1024) */, nullptr);

  // 3) softmax along key axis, in place
  softmax_rows<<<dim3(BATCH * SEQ), 256, 0, stream>>>(Pb, SEQ);

  // 4) attn_out = P @ V^T  (M=2048, N=1024, K=2048; V rows are features, contiguous in t)
  gemm_bf16_tt<1><<<dim3(DIM / 128, SEQ / 64, BATCH), 256, 0, stream>>>(
      Pb, SEQ, (long)SEQ * SEQ,
      Vb, SEQ, (long)DIM * SEQ,
      AOb, DIM, (long)SEQ * DIM,
      SEQ, 1.0f, nullptr);

  // 5) out = attn_out @ W^T + b  (batch folded: M = B*S = 8192, N=E=1024, K=D=1024)
  gemm_bf16_tt<2><<<dim3(EMB / 128, (BATCH * SEQ) / 64, 1), 256, 0, stream>>>(
      AOb, DIM, 0L,
      Wb,  DIM, 0L,
      d_out, EMB, 0L,
      DIM, 1.0f, b);
}
